// NN_obj_layer_51642686767182
// MI455X (gfx1250) — matrix operand order repaired and checked by reading, not compiled, not run on hardware
//
#include <hip/hip_runtime.h>
#include <hip/hip_bf16.h>

#define STEPS 96
#define NSAMP 2048
#define TROWS (NSAMP * STEPS)

typedef __attribute__((ext_vector_type(16))) _Float16 v16h;
typedef __attribute__((ext_vector_type(8)))  float    v8f;

// ---------------- WMMA fragment helpers (CDNA5 16x16x32 f16 -> f32) --------

__device__ __forceinline__ v16h load_A_sm(const _Float16* sm, int ld, int k0, int lane) {
    // A: 16x32 (MxK). lanes 0-15: rows 0-15, K = {0..7,16..23}; lanes 16-31: K = {8..15,24..31}
    int row = lane & 15;
    int kb  = (lane & 16) ? 8 : 0;
    const _Float16* p = sm + row * ld + k0 + kb;
    v16h a;
#pragma unroll
    for (int i = 0; i < 8; ++i) { a[i] = p[i]; a[8 + i] = p[16 + i]; }
    return a;
}

// B fragments are pre-packed fragment-major: 32 contiguous bytes per lane.
__device__ __forceinline__ v16h load_B_pk(const _Float16* P, int frag, int lane) {
    const _Float16* p = P + ((size_t)frag * 32 + lane) * 16;
    v16h b;
#pragma unroll
    for (int i = 0; i < 16; ++i) b[i] = p[i];
    return b;
}

__device__ __forceinline__ v8f wmma16(v16h a, v16h b, v8f c) {
    return __builtin_amdgcn_wmma_f32_16x16x32_f16(
        /*neg_a=*/false, a, /*neg_b=*/false, b,
        /*c_mod=*/(short)0, c, /*reuse_a=*/false, /*reuse_b=*/false);
}

__device__ __forceinline__ void store_relu_sm(_Float16* sm, int ld, int n0, v8f c,
                                              const float* bias, int lane) {
    // C/D: lanes 0-15 rows 0..7 per vgpr r; lanes 16-31 rows 8..15
    int col = lane & 15;
    int rb  = (lane & 16) ? 8 : 0;
    float bv = bias[n0 + col];
#pragma unroll
    for (int r = 0; r < 8; ++r) {
        float v = c[r] + bv;
        v = fmaxf(v, 0.f);
        sm[(rb + r) * ld + n0 + col] = (_Float16)v;
    }
}

// ---------------- weight pack: f32 [K][N] -> f16 WMMA fragments ------------
// Fragment f = kchunk * (Npad/16) + tile; element layout [frag][lane][i].
// lane: col = lane&15, kb = (lane&16)?8:0; element i -> K = kc*32+kb+(i&7)+((i&8)?16:0).

__global__ void pack_w(const float* __restrict__ src, _Float16* __restrict__ dst,
                       int K, int N, int Kpad, int Npad) {
    int e = blockIdx.x * blockDim.x + threadIdx.x;
    if (e >= Kpad * Npad) return;
    int ntl  = Npad >> 4;
    int f    = e >> 9;        // 512 halves per fragment
    int r    = e & 511;
    int lane = r >> 4;
    int i    = r & 15;
    int kc   = f / ntl, tile = f % ntl;
    int col  = lane & 15;
    int kb   = (lane & 16) ? 8 : 0;
    int k    = kc * 32 + kb + (i & 7) + ((i & 8) ? 16 : 0);
    int n    = tile * 16 + col;
    float v  = (k < K && n < N) ? src[k * N + n] : 0.f;
    dst[e] = (_Float16)v;
}

__global__ void pad_bias(const float* __restrict__ src, float* __restrict__ dst,
                         int N, int Npad) {
    int i = blockIdx.x * blockDim.x + threadIdx.x;
    if (i >= Npad) return;
    dst[i] = (i < N) ? src[i] : 0.f;
}

__global__ void init_u0(const float* __restrict__ enY, float* __restrict__ u_seq) {
    int i = blockIdx.x * blockDim.x + threadIdx.x;
    if (i >= NSAMP * 10) return;
    int n = i / 10, d = i % 10;
    size_t off = (size_t)n * STEPS * 10 + d;    // row n*STEPS + step 0
    u_seq[off] = enY[off];
}

// ---------------- recurrent step: 26 -> 200 -> 200 -> 10 ------------------
// One launch per scan step; block = 16 samples, 128 threads = 4 waves.
// Padded: L1 K=32,N=208 (ntl=13); L2 K=224,N=208; L3 K=224,N=16 (ntl=1).

__global__ void __launch_bounds__(128)
mm_step_kernel(float* __restrict__ u_seq,
               const float* __restrict__ enP, const float* __restrict__ enX,
               const _Float16* __restrict__ W1, const float* __restrict__ b1,
               const _Float16* __restrict__ W2, const float* __restrict__ b2,
               const _Float16* __restrict__ W3, const float* __restrict__ b3,
               int j) {
    __shared__ _Float16 sx [16 * 32];
    __shared__ _Float16 sh1[16 * 224];
    __shared__ _Float16 sh2[16 * 224];

    int tid = threadIdx.x, lane = tid & 31, wave = tid >> 5;
    int s0 = blockIdx.x * 16;

    // stage x = [u | pP | pX], zero-padded to 32
    for (int i = tid; i < 16 * 32; i += 128) {
        int r = i >> 5, c = i & 31;
        size_t t = (size_t)(s0 + r) * STEPS + j;
        float v = 0.f;
        if (c < 10)      v = u_seq[t * 10 + c];
        else if (c < 18) v = enP[t * 8 + (c - 10)];
        else if (c < 26) v = enX[t * 8 + (c - 18)];
        sx[r * 32 + c] = (_Float16)v;
    }
    // zero pad cols 208..223 of h1/h2 (never written by the 13 N-tiles)
    for (int i = tid; i < 16 * 16; i += 128) {
        int r = i >> 4, c = 208 + (i & 15);
        sh1[r * 224 + c] = (_Float16)0.f;
        sh2[r * 224 + c] = (_Float16)0.f;
    }
    __syncthreads();

    // layer 1: K=32 (1 chunk), N=208 (13 tiles)
    for (int tile = wave; tile < 13; tile += 4) {
        v8f c = {};
        v16h a = load_A_sm(sx, 32, 0, lane);
        v16h b = load_B_pk(W1, tile, lane);          // kc=0, ntl=13
        c = wmma16(a, b, c);
        store_relu_sm(sh1, 224, tile * 16, c, b1, lane);
    }
    __syncthreads();

    // layer 2: K=224 (7 chunks), N=208
    for (int tile = wave; tile < 13; tile += 4) {
        v8f c = {};
#pragma unroll
        for (int kc = 0; kc < 7; ++kc) {
            v16h a = load_A_sm(sh1, 224, kc * 32, lane);
            v16h b = load_B_pk(W2, kc * 13 + tile, lane);
            c = wmma16(a, b, c);
        }
        store_relu_sm(sh2, 224, tile * 16, c, b2, lane);
    }
    __syncthreads();

    // layer 3: K=224 (7 chunks), N=16 (one tile, wave 0); write u_{j+1}
    if (wave == 0) {
        v8f c = {};
#pragma unroll
        for (int kc = 0; kc < 7; ++kc) {
            v16h a = load_A_sm(sh2, 224, kc * 32, lane);
            v16h b = load_B_pk(W3, kc, lane);        // ntl=1
            c = wmma16(a, b, c);
        }
        int col = lane & 15;
        int rb  = (lane & 16) ? 8 : 0;
        if (col < 10) {
            float bv = b3[col];
#pragma unroll
            for (int r = 0; r < 8; ++r) {
                size_t t = (size_t)(s0 + rb + r) * STEPS + j + 1;
                u_seq[t * 10 + col] = c[r] + bv;
            }
        }
    }
}

// ---------------- fused decoder (DIN -> 64 -> 128 -> 256 -> NOUT) ---------
// block = 16 rows, 128 threads = 4 waves; activations ping-pong in LDS.
// Fused epilogue writes decoder output + objective arrays.

template <int DIN, int NOUT, int NOUTP, bool HASP>
__global__ void __launch_bounds__(128)
decoder_kernel(const float* __restrict__ xsrc,
               const _Float16* __restrict__ W1, const float* __restrict__ b1,
               const _Float16* __restrict__ W2, const float* __restrict__ b2,
               const _Float16* __restrict__ W3, const float* __restrict__ b3,
               const _Float16* __restrict__ W4, const float* __restrict__ b4,
               const float* __restrict__ scale, const float* __restrict__ sbias,
               const float* __restrict__ wei,   const float* __restrict__ price,
               const float* __restrict__ Upp,   const float* __restrict__ Low,
               float* __restrict__ outA, float* __restrict__ outB,
               float* __restrict__ outP1, float* __restrict__ outDe) {
    __shared__ _Float16 smA[16 * 256];
    __shared__ _Float16 smB[16 * 256];

    int tid = threadIdx.x, lane = tid & 31, wave = tid >> 5;
    int t0 = blockIdx.x * 16;

    // stage x (zero-padded to K=32) into smA (ld 32)
    for (int i = tid; i < 16 * 32; i += 128) {
        int r = i >> 5, c = i & 31;
        float v = (c < DIN) ? xsrc[(size_t)(t0 + r) * DIN + c] : 0.f;
        smA[r * 32 + c] = (_Float16)v;
    }
    __syncthreads();

    // L1: K=32, N=64 (4 tiles, ntl=4) -> smB ld 64
    {
        v8f c = {};
        v16h a = load_A_sm(smA, 32, 0, lane);
        v16h b = load_B_pk(W1, wave, lane);
        c = wmma16(a, b, c);
        store_relu_sm(smB, 64, wave * 16, c, b1, lane);
    }
    __syncthreads();

    // L2: K=64 (2 chunks), N=128 (8 tiles, ntl=8) -> smA ld 128
#pragma unroll
    for (int tt = 0; tt < 2; ++tt) {
        int tile = wave + tt * 4;
        v8f c = {};
#pragma unroll
        for (int kc = 0; kc < 2; ++kc) {
            v16h a = load_A_sm(smB, 64, kc * 32, lane);
            v16h b = load_B_pk(W2, kc * 8 + tile, lane);
            c = wmma16(a, b, c);
        }
        store_relu_sm(smA, 128, tile * 16, c, b2, lane);
    }
    __syncthreads();

    // L3: K=128 (4 chunks), N=256 (16 tiles, ntl=16) -> smB ld 256
#pragma unroll
    for (int tt = 0; tt < 4; ++tt) {
        int tile = wave + tt * 4;
        v8f c = {};
#pragma unroll
        for (int kc = 0; kc < 4; ++kc) {
            v16h a = load_A_sm(smA, 128, kc * 32, lane);
            v16h b = load_B_pk(W3, kc * 16 + tile, lane);
            c = wmma16(a, b, c);
        }
        store_relu_sm(smB, 256, tile * 16, c, b3, lane);
    }
    __syncthreads();

    // L4: K=256 (8 chunks), N=NOUTP tiles (ntl=NOUTP/16); fused epilogue
    const int ntiles = NOUTP / 16;
    float w = wei[0];
    for (int tile = wave; tile < ntiles; tile += 4) {
        v8f c = {};
#pragma unroll
        for (int kc = 0; kc < 8; ++kc) {
            v16h a = load_A_sm(smB, 256, kc * 32, lane);
            v16h b = load_B_pk(W4, kc * ntiles + tile, lane);
            c = wmma16(a, b, c);
        }
        int col = lane & 15;
        int n   = tile * 16 + col;
        int rb  = (lane & 16) ? 8 : 0;
        if (n < NOUT) {
            float bv = b4[n], sc = scale[n], sb = sbias[n];
#pragma unroll
            for (int r = 0; r < 8; ++r) {
                size_t t   = (size_t)(t0 + rb + r);
                size_t idx = t * NOUT + n;
                float de = c[r] + bv;
                outDe[idx] = de;
                float hy = de * sc + sb;
                if (HASP) outP1[idx] = hy * price[idx];
                outA[idx] = fmaxf(hy * w - Upp[idx], 0.f);
                outB[idx] = fmaxf(Low[idx] - hy * w, 0.f);
            }
        }
    }
}

// -------------------------------- host side -------------------------------

extern "C" void kernel_launch(void* const* d_in, const int* in_sizes, int n_in,
                              void* d_out, int out_size, void* d_ws, size_t ws_size,
                              hipStream_t stream) {
    (void)in_sizes; (void)n_in; (void)out_size; (void)ws_size;
    const float* enY    = (const float*)d_in[0];
    const float* enP    = (const float*)d_in[1];
    const float* enX    = (const float*)d_in[2];
    const float* Pprice = (const float*)d_in[3];
    const float* UppY   = (const float*)d_in[4];
    const float* LowY   = (const float*)d_in[5];
    const float* UppP   = (const float*)d_in[6];
    const float* LowP   = (const float*)d_in[7];
    const float* scaleY = (const float*)d_in[30];
    const float* biasY  = (const float*)d_in[31];
    const float* scaleP = (const float*)d_in[32];
    const float* biasP  = (const float*)d_in[33];
    const float* weiY   = (const float*)d_in[34];
    const float* weiP   = (const float*)d_in[35];

    char* ws = (char*)d_ws;
    size_t off = 0;
    auto alloc = [&](size_t bytes) { off = (off + 255) & ~(size_t)255; size_t o = off; off += bytes; return o; };

    float* u_seq = (float*)(ws + alloc((size_t)TROWS * 10 * 4));

    // packed f16 weight fragments: {input idx, K, N, Kpad, Npad}
    struct WD { int idx, K, N, Kpad, Npad; size_t o; };
    WD wd[11] = {
        { 8,  26, 200,  32, 208, 0}, {10, 200, 200, 224, 208, 0}, {12, 200,  10, 224,  16, 0},
        {14,  10,  64,  32,  64, 0}, {16,  64, 128,  64, 128, 0}, {18, 128, 256, 128, 256, 0},
        {20, 256,  90, 256,  96, 0},
        {22,   8,  64,  32,  64, 0}, {24,  64, 128,  64, 128, 0}, {26, 128, 256, 128, 256, 0},
        {28, 256,  80, 256,  80, 0},
    };
    for (auto& w : wd) w.o = alloc((size_t)w.Kpad * w.Npad * 2);

    // padded f32 biases: {input idx, N, Npad}
    struct BD { int idx, N, Npad; size_t o; };
    BD bd[11] = {
        { 9, 200, 208, 0}, {11, 200, 208, 0}, {13,  10,  16, 0},
        {15,  64,  64, 0}, {17, 128, 128, 0}, {19, 256, 256, 0}, {21, 90, 96, 0},
        {23,  64,  64, 0}, {25, 128, 128, 0}, {27, 256, 256, 0}, {29, 80, 80, 0},
    };
    for (auto& b : bd) b.o = alloc((size_t)b.Npad * 4);

    for (auto& w : wd) {
        int tot = w.Kpad * w.Npad;
        pack_w<<<(tot + 255) / 256, 256, 0, stream>>>(
            (const float*)d_in[w.idx], (_Float16*)(ws + w.o), w.K, w.N, w.Kpad, w.Npad);
    }
    for (auto& b : bd) {
        pad_bias<<<(b.Npad + 255) / 256, 256, 0, stream>>>(
            (const float*)d_in[b.idx], (float*)(ws + b.o), b.N, b.Npad);
    }

    // u_seq step 0 = en_Y step 0
    init_u0<<<(NSAMP * 10 + 255) / 256, 256, 0, stream>>>(enY, u_seq);

    const _Float16* mmW1 = (const _Float16*)(ws + wd[0].o);
    const _Float16* mmW2 = (const _Float16*)(ws + wd[1].o);
    const _Float16* mmW3 = (const _Float16*)(ws + wd[2].o);
    const float* mmb1 = (const float*)(ws + bd[0].o);
    const float* mmb2 = (const float*)(ws + bd[1].o);
    const float* mmb3 = (const float*)(ws + bd[2].o);

    // sequential scan: 95 dependent steps, 128 blocks x 16 samples each
    for (int j = 0; j < STEPS - 1; ++j) {
        mm_step_kernel<<<NSAMP / 16, 128, 0, stream>>>(
            u_seq, enP, enX, mmW1, mmb1, mmW2, mmb2, mmW3, mmb3, j);
    }

    // output layout (flat, return order)
    float* out = (float*)d_out;
    size_t o_upY = 0;
    size_t o_loY = (size_t)TROWS * 90;
    size_t o_P1  = o_loY + (size_t)TROWS * 90;
    size_t o_P2u = o_P1  + (size_t)TROWS * 80;
    size_t o_P2l = o_P2u + (size_t)TROWS * 80;
    size_t o_deP = o_P2l + (size_t)TROWS * 80;
    size_t o_deY = o_deP + (size_t)TROWS * 80;

    decoder_kernel<10, 90, 96, false><<<TROWS / 16, 128, 0, stream>>>(
        u_seq,
        (const _Float16*)(ws + wd[3].o), (const float*)(ws + bd[3].o),
        (const _Float16*)(ws + wd[4].o), (const float*)(ws + bd[4].o),
        (const _Float16*)(ws + wd[5].o), (const float*)(ws + bd[5].o),
        (const _Float16*)(ws + wd[6].o), (const float*)(ws + bd[6].o),
        scaleY, biasY, weiY, (const float*)nullptr, UppY, LowY,
        out + o_upY, out + o_loY, (float*)nullptr, out + o_deY);

    decoder_kernel<8, 80, 80, true><<<TROWS / 16, 128, 0, stream>>>(
        enP,
        (const _Float16*)(ws + wd[7].o),  (const float*)(ws + bd[7].o),
        (const _Float16*)(ws + wd[8].o),  (const float*)(ws + bd[8].o),
        (const _Float16*)(ws + wd[9].o),  (const float*)(ws + bd[9].o),
        (const _Float16*)(ws + wd[10].o), (const float*)(ws + bd[10].o),
        scaleP, biasP, weiP, Pprice, UppP, LowP,
        out + o_P2u, out + o_P2l, out + o_P1, out + o_deP);
}
